// LlamaAttention_34849364639782
// MI455X (gfx1250) — matrix operand order repaired and checked by reading, not compiled, not run on hardware
//
#include <hip/hip_runtime.h>
#include <cstdint>
#include <cstddef>

// ---------- constants ----------
#define B_ 2
#define S_ 2048
#define HID_ 1536
#define H_ 12
#define HD_ 128
#define LAT_ 192
#define ROT_ 64
#define T_ (B_ * S_)   // 4096 tokens

typedef __attribute__((ext_vector_type(16))) __bf16 v16bf;
typedef __attribute__((ext_vector_type(8)))  float  v8f;

union Frag {
    v16bf v;
    unsigned int d[8];
    unsigned short h[16];
};

__device__ __forceinline__ unsigned short f2bf(float x) {
    union { float f; unsigned int u; } c; c.f = x;
    unsigned int u = c.u;
    unsigned int r = u + 0x7fffu + ((u >> 16) & 1u);   // round-to-nearest-even
    return (unsigned short)(r >> 16);
}
__device__ __forceinline__ float bf2f(unsigned short b) {
    union { unsigned int u; float f; } c; c.u = ((unsigned int)b) << 16;
    return c.f;
}

// ---------- f32 -> bf16 conversion ----------
__global__ void __launch_bounds__(256) conv_bf16_kernel(const float* __restrict__ x,
                                                        unsigned short* __restrict__ y, int n) {
    int i = blockIdx.x * 256 + threadIdx.x;
    if (i < n) y[i] = f2bf(x[i]);
}

// ---------- generic bf16 WMMA GEMM: C(M,N) = A(M,K) @ B(K,N) ----------
// A, B row-major bf16. Block tile 64x64, BK=32, 256 threads (8 waves).
// M % 64 == 0, N % 64 == 0, K % 32 == 0 (true for all calls here).
template <bool OUT_F32>
__global__ void __launch_bounds__(256, 1)
gemm_bf16_wmma(const unsigned short* __restrict__ A,
               const unsigned short* __restrict__ Bm,
               void* __restrict__ Cp,
               int M, int N, int K) {
    __shared__ unsigned short As[64 * 32];   // [m][k]
    __shared__ unsigned short Bs[64 * 32];   // [n][k] (transposed on stage)

    const int tid  = threadIdx.x;
    const int wave = tid >> 5;
    const int lane = tid & 31;
    const int lr   = lane & 15;
    const int hi   = lane >> 4;
    const int m0   = blockIdx.y * 64;
    const int n0   = blockIdx.x * 64;
    const int mt   = wave >> 1;          // 0..3
    const int nt0  = (wave & 1) * 2;     // 0 or 2

    v8f acc0 = {}; v8f acc1 = {};

    for (int k0 = 0; k0 < K; k0 += 32) {
        // stage A tile (64x32 bf16), 8 elems/thread, contiguous k in global
        #pragma unroll
        for (int j = 0; j < 8; ++j) {
            int idx = tid + 256 * j;           // 0..2047
            int kk = idx & 31, mm = idx >> 5;
            As[mm * 32 + kk] = A[(size_t)(m0 + mm) * K + k0 + kk];
        }
        // stage B tile transposed: Bs[n][k] = B[k0+k][n0+n]; coalesced on n
        #pragma unroll
        for (int j = 0; j < 8; ++j) {
            int idx = tid + 256 * j;
            int nn = idx & 63, kk = idx >> 6;
            Bs[nn * 32 + kk] = Bm[(size_t)(k0 + kk) * N + n0 + nn];
        }
        __syncthreads();

        // A fragment: lane = row lr (of 16), element i -> k = i + 16*hi
        Frag af;
        {
            const unsigned int* Au = (const unsigned int*)As;
            int row = mt * 16 + lr;
            #pragma unroll
            for (int p = 0; p < 8; ++p) {
                int k = 2 * p + 16 * hi;
                af.d[p] = Au[row * 16 + (k >> 1)];
            }
        }
        // B fragments: lane = col, elements = 16 contiguous k starting at 16*hi
        Frag bf0, bf1;
        {
            const unsigned int* Bu = (const unsigned int*)Bs;
            int c0 = (nt0 + 0) * 16 + lr;
            int c1 = (nt0 + 1) * 16 + lr;
            int kb = 8 * hi;                   // dword offset
            #pragma unroll
            for (int p = 0; p < 8; ++p) {
                bf0.d[p] = Bu[c0 * 16 + kb + p];
                bf1.d[p] = Bu[c1 * 16 + kb + p];
            }
        }
        acc0 = __builtin_amdgcn_wmma_f32_16x16x32_bf16(false, af.v, false, bf0.v,
                                                       (short)0, acc0, false, false);
        acc1 = __builtin_amdgcn_wmma_f32_16x16x32_bf16(false, af.v, false, bf1.v,
                                                       (short)0, acc1, false, false);
        __syncthreads();
    }

    // C/D layout: lane lr = col, VGPR v -> row = v + 8*hi
    int rbase = m0 + mt * 16 + 8 * hi;
    int c0 = n0 + (nt0 + 0) * 16 + lr;
    int c1 = n0 + (nt0 + 1) * 16 + lr;
    #pragma unroll
    for (int v = 0; v < 8; ++v) {
        size_t r = (size_t)(rbase + v);
        if (OUT_F32) {
            float* C = (float*)Cp;
            C[r * N + c0] = acc0[v];
            C[r * N + c1] = acc1[v];
        } else {
            unsigned short* C = (unsigned short*)Cp;
            C[r * N + c0] = f2bf(acc0[v]);
            C[r * N + c1] = f2bf(acc1[v]);
        }
    }
}

// ---------- RoPE + concat: out(t, h, 0:64)=xc, out(t, h, 64:128)=rope(xr) ----------
__global__ void __launch_bounds__(256) rope_concat_kernel(
        const unsigned short* __restrict__ xc,  // (T, H*64)
        const unsigned short* __restrict__ xr,  // (T, H*64)
        const float* __restrict__ emb,          // (S, 64)
        unsigned short* __restrict__ out) {     // (T, H*128)
    int g = blockIdx.x * 256 + threadIdx.x;    // T_*H_ = 49152
    int h = g % H_;
    int t = g / H_;
    int s = t % S_;
    const unsigned short* c = xc + (size_t)t * (H_ * ROT_) + h * ROT_;
    const unsigned short* r = xr + (size_t)t * (H_ * ROT_) + h * ROT_;
    unsigned short* o = out + (size_t)t * HID_ + h * HD_;
    #pragma unroll 8
    for (int j = 0; j < 64; ++j) o[j] = c[j];
    const float* e = emb + (size_t)s * ROT_;
    #pragma unroll 4
    for (int j = 0; j < 32; ++j) {
        float x1 = bf2f(r[j]), x2 = bf2f(r[j + 32]);
        float sn = e[j], cs = e[j + 32];
        o[64 + j] = f2bf(x1 * cs - x2 * sn);
        o[96 + j] = f2bf(x1 * sn + x2 * cs);
    }
}

// ---------- fused causal flash attention (bf16 in, bf16 out) ----------
// grid.x = B_*H_ (24), grid.y = S_/64 (32), block = 128 (4 waves, 16 q-rows each).
// __launch_bounds__(128, 1): allow full VGPR allocation, no scratch spills in the
// key-block loop (acc 64 + qf 32 + scores 16 + stats 24 + frags ~32 ≈ 190 VGPRs).
__global__ void __launch_bounds__(128, 1)
mla_flash_attn_kernel(const unsigned short* __restrict__ Q,   // (T,1536)
                      const unsigned short* __restrict__ Kg,  // (T,1536)
                      const unsigned short* __restrict__ Vg,  // (T,1536)
                      unsigned short* __restrict__ O) {       // (T,1536)
    __shared__ unsigned short Ks[32 * 128];       // [key][d]
    __shared__ unsigned short Vs[128 * 32];       // [d][key]   (transposed)
    __shared__ unsigned short Ps[4 * 16 * 32];    // per-wave P tile [row][key]

    const int tid  = threadIdx.x;
    const int wave = tid >> 5;
    const int lane = tid & 31;
    const int lr   = lane & 15;
    const int hi   = lane >> 4;
    const int b    = blockIdx.x / H_;
    const int h    = blockIdx.x % H_;
    const int qwg  = blockIdx.y * 64;
    const int qbase = qwg + wave * 16;
    const size_t tok0 = (size_t)b * S_;

    // Q A-fragments for 4 d-chunks of 32 (loaded once, stay in VGPRs)
    Frag qf[4];
    {
        const unsigned short* qp = Q + (tok0 + qbase + lr) * HID_ + h * HD_;
        #pragma unroll
        for (int dc = 0; dc < 4; ++dc) {
            #pragma unroll
            for (int p = 0; p < 8; ++p) {
                int k = 2 * p + 16 * hi;
                qf[dc].d[p] = *(const unsigned int*)(qp + dc * 32 + k);
            }
        }
    }

    v8f acc[8];
    #pragma unroll
    for (int c = 0; c < 8; ++c) acc[c] = (v8f){0, 0, 0, 0, 0, 0, 0, 0};
    float rowM[8], rowL[8];
    #pragma unroll
    for (int v = 0; v < 8; ++v) { rowM[v] = -1e30f; rowL[v] = 0.0f; }

    const float scale = 0.08838834764831845f;   // 1/sqrt(128)
    const int nkb = qwg / 32 + 2;               // key blocks of 32 covering causal extent
    const int qrow0 = qbase + 8 * hi;           // per-lane row base (row = qrow0 + v)

    for (int kb = 0; kb < nkb; ++kb) {
        __syncthreads();
        // stage K tile (32 keys x 128 d) as dwords, 16/thread
        {
            const unsigned int* src =
                (const unsigned int*)(Kg + (tok0 + (size_t)kb * 32) * HID_ + h * HD_);
            unsigned int* dst = (unsigned int*)Ks;
            #pragma unroll
            for (int j = 0; j < 16; ++j) {
                int idx = tid + 128 * j;         // 0..2047 dwords
                int key = idx >> 6, dd = idx & 63;
                dst[key * 64 + dd] = src[(size_t)key * (HID_ / 2) + dd];
            }
        }
        // stage V transposed: Vs[d][key]
        {
            const unsigned short* src = Vg + (tok0 + (size_t)kb * 32) * HID_ + h * HD_;
            int d = tid;                         // 0..127
            #pragma unroll
            for (int key = 0; key < 32; ++key)
                Vs[d * 32 + key] = src[(size_t)key * HID_ + d];
        }
        __syncthreads();

        // scores: S[16q x 32k] = Q(16x128) . K^T, two 16-col tiles
        v8f s0 = (v8f){0, 0, 0, 0, 0, 0, 0, 0};
        v8f s1 = (v8f){0, 0, 0, 0, 0, 0, 0, 0};
        const unsigned int* Ku = (const unsigned int*)Ks;
        #pragma unroll
        for (int dc = 0; dc < 4; ++dc) {
            Frag b0, b1;
            int kd = dc * 16 + 8 * hi;           // dword offset in 64-dword key row
            #pragma unroll
            for (int p = 0; p < 8; ++p) {
                b0.d[p] = Ku[(lr + 0)  * 64 + kd + p];
                b1.d[p] = Ku[(lr + 16) * 64 + kd + p];
            }
            s0 = __builtin_amdgcn_wmma_f32_16x16x32_bf16(false, qf[dc].v, false, b0.v,
                                                         (short)0, s0, false, false);
            s1 = __builtin_amdgcn_wmma_f32_16x16x32_bf16(false, qf[dc].v, false, b1.v,
                                                         (short)0, s1, false, false);
        }

        // online softmax (rows qrow0+v per lane; reduce over 16-lane half-groups)
        const int kg0 = kb * 32 + lr;            // this lane's key column (tile 0)
        const int kg1 = kg0 + 16;                // tile 1
        float f[8];
        #pragma unroll
        for (int v = 0; v < 8; ++v) {
            const int qg = qrow0 + v;
            float a = (kg0 > qg) ? -1e30f : s0[v] * scale;
            float c = (kg1 > qg) ? -1e30f : s1[v] * scale;
            float t = fmaxf(a, c);
            t = fmaxf(t, __shfl_xor(t, 1, 32));
            t = fmaxf(t, __shfl_xor(t, 2, 32));
            t = fmaxf(t, __shfl_xor(t, 4, 32));
            t = fmaxf(t, __shfl_xor(t, 8, 32));
            float nm = fmaxf(rowM[v], t);
            float fv = __expf(rowM[v] - nm);
            rowM[v] = nm;
            float p0 = __expf(a - nm);
            float p1 = __expf(c - nm);
            int row = v + 8 * hi;
            Ps[wave * 512 + row * 32 + lr]      = f2bf(p0);
            Ps[wave * 512 + row * 32 + 16 + lr] = f2bf(p1);
            float ps = p0 + p1;
            ps += __shfl_xor(ps, 1, 32);
            ps += __shfl_xor(ps, 2, 32);
            ps += __shfl_xor(ps, 4, 32);
            ps += __shfl_xor(ps, 8, 32);
            rowL[v] = rowL[v] * fv + ps;
            f[v] = fv;
        }
        #pragma unroll
        for (int c = 0; c < 8; ++c)
            #pragma unroll
            for (int v = 0; v < 8; ++v) acc[c][v] *= f[v];

        // LDS stores above feed same-wave loads below (DS in-order; fence compiler)
        asm volatile("s_wait_dscnt 0" ::: "memory");

        // P A-fragment (16q x 32k)
        Frag pf;
        {
            const unsigned int* Pu = (const unsigned int*)(Ps + wave * 512);
            #pragma unroll
            for (int p = 0; p < 8; ++p) {
                int k = 2 * p + 16 * hi;
                pf.d[p] = Pu[lr * 16 + (k >> 1)];
            }
        }
        // acc += P(16x32) . V(32x128), 8 n-chunks of 16
        const unsigned int* Vu = (const unsigned int*)Vs;
        #pragma unroll
        for (int c = 0; c < 8; ++c) {
            Frag vf;
            int dcol = c * 16 + lr;
            #pragma unroll
            for (int p = 0; p < 8; ++p)
                vf.d[p] = Vu[dcol * 16 + 8 * hi + p];
            acc[c] = __builtin_amdgcn_wmma_f32_16x16x32_bf16(false, pf.v, false, vf.v,
                                                             (short)0, acc[c], false, false);
        }
    }

    // normalize and write attn (bf16)
    unsigned short* op = O + (tok0 + qbase) * HID_ + h * HD_;
    #pragma unroll
    for (int v = 0; v < 8; ++v) {
        float inv = 1.0f / rowL[v];
        int row = v + 8 * hi;
        #pragma unroll
        for (int c = 0; c < 8; ++c)
            op[(size_t)row * HID_ + c * 16 + lr] = f2bf(acc[c][v] * inv);
    }
}

// ---------- host ----------
extern "C" void kernel_launch(void* const* d_in, const int* in_sizes, int n_in,
                              void* d_out, int out_size, void* d_ws, size_t ws_size,
                              hipStream_t stream) {
    (void)in_sizes; (void)n_in; (void)out_size; (void)ws_size;

    const float* hid    = (const float*)d_in[0];
    const float* rot    = (const float*)d_in[1];   // (1,1,S,64)
    const float* wkvd   = (const float*)d_in[2];   // (1536,192)
    const float* wqd    = (const float*)d_in[3];   // (1536,192)
    const float* wku    = (const float*)d_in[4];   // (192,768)
    const float* wqu    = (const float*)d_in[5];   // (192,768)
    const float* wvu    = (const float*)d_in[6];   // (192,1536)
    const float* wropek = (const float*)d_in[7];   // (1536,768)
    const float* wropeq = (const float*)d_in[8];   // (192,768)
    const float* wo     = (const float*)d_in[9];   // (1536,1536)

    size_t off = 0;
    auto alloc = [&](size_t elems) -> unsigned short* {
        unsigned short* p = (unsigned short*)((char*)d_ws + off);
        off += ((elems * sizeof(unsigned short) + 255) & ~(size_t)255);
        return p;
    };

    unsigned short* hbf   = alloc((size_t)T_ * HID_);
    unsigned short* bkvd  = alloc((size_t)HID_ * LAT_);
    unsigned short* bqd   = alloc((size_t)HID_ * LAT_);
    unsigned short* bku   = alloc((size_t)LAT_ * (H_ * ROT_));
    unsigned short* bqu   = alloc((size_t)LAT_ * (H_ * ROT_));
    unsigned short* bvu   = alloc((size_t)LAT_ * HID_);
    unsigned short* bropk = alloc((size_t)HID_ * (H_ * ROT_));
    unsigned short* bropq = alloc((size_t)LAT_ * (H_ * ROT_));
    unsigned short* bwo   = alloc((size_t)HID_ * HID_);
    unsigned short* kvd   = alloc((size_t)T_ * LAT_);
    unsigned short* qd    = alloc((size_t)T_ * LAT_);
    unsigned short* kc    = alloc((size_t)T_ * (H_ * ROT_));
    unsigned short* qc    = alloc((size_t)T_ * (H_ * ROT_));
    unsigned short* kr    = alloc((size_t)T_ * (H_ * ROT_));
    unsigned short* qr    = alloc((size_t)T_ * (H_ * ROT_));
    unsigned short* Qb    = alloc((size_t)T_ * HID_);
    unsigned short* Kb    = alloc((size_t)T_ * HID_);
    unsigned short* Vb    = alloc((size_t)T_ * HID_);
    unsigned short* attn  = alloc((size_t)T_ * HID_);

    auto conv = [&](const float* s, unsigned short* dst, size_t n) {
        conv_bf16_kernel<<<(unsigned)((n + 255) / 256), 256, 0, stream>>>(s, dst, (int)n);
    };
    conv(hid,    hbf,   (size_t)T_ * HID_);
    conv(wkvd,   bkvd,  (size_t)HID_ * LAT_);
    conv(wqd,    bqd,   (size_t)HID_ * LAT_);
    conv(wku,    bku,   (size_t)LAT_ * (H_ * ROT_));
    conv(wqu,    bqu,   (size_t)LAT_ * (H_ * ROT_));
    conv(wvu,    bvu,   (size_t)LAT_ * HID_);
    conv(wropek, bropk, (size_t)HID_ * (H_ * ROT_));
    conv(wropeq, bropq, (size_t)LAT_ * (H_ * ROT_));
    conv(wo,     bwo,   (size_t)HID_ * HID_);

    auto gemm_bf = [&](const unsigned short* A, const unsigned short* Bm,
                       unsigned short* C, int M, int N, int K) {
        dim3 g(N / 64, M / 64);
        gemm_bf16_wmma<false><<<g, 256, 0, stream>>>(A, Bm, (void*)C, M, N, K);
    };

    gemm_bf(hbf, bkvd, kvd, T_, LAT_, HID_);                 // kv_d
    gemm_bf(hbf, bqd,  qd,  T_, LAT_, HID_);                 // q_d
    gemm_bf(kvd, bku,  kc,  T_, H_ * ROT_, LAT_);            // k_c
    gemm_bf(qd,  bqu,  qc,  T_, H_ * ROT_, LAT_);            // q_c
    gemm_bf(kvd, bvu,  Vb,  T_, HID_, LAT_);                 // v
    gemm_bf(hbf, bropk, kr, T_, H_ * ROT_, HID_);            // k_r
    gemm_bf(qd,  bropq, qr, T_, H_ * ROT_, LAT_);            // q_r

    rope_concat_kernel<<<(T_ * H_) / 256, 256, 0, stream>>>(qc, qr, rot, Qb);
    rope_concat_kernel<<<(T_ * H_) / 256, 256, 0, stream>>>(kc, kr, rot, Kb);

    mla_flash_attn_kernel<<<dim3(B_ * H_, S_ / 64), 128, 0, stream>>>(Qb, Kb, Vb, attn);

    dim3 go(HID_ / 64, T_ / 64);
    gemm_bf16_wmma<true><<<go, 256, 0, stream>>>(attn, bwo, d_out, T_, HID_, HID_);
}
